// CausalTemporalAttention_51539607552834
// MI455X (gfx1250) — matrix operand order repaired and checked by reading, not compiled, not run on hardware
//
#include <hip/hip_runtime.h>

// ---------------------------------------------------------------------------
// CausalTemporalAttention for MI455X (gfx1250), wave32 + WMMA bf16 +
// async global->LDS double buffering.  B=2, S=2048, D=1024, H=16, HD=64.
// ---------------------------------------------------------------------------

typedef __bf16 v16bf __attribute__((ext_vector_type(16)));
typedef float  v8f   __attribute__((ext_vector_type(8)));

#define S_LEN  2048
#define DMODEL 1024
#define NHEAD  16
#define HDIM   64
#define BATCH  2

__device__ __forceinline__ __bf16 f2bf(float f) {
  // round-to-nearest-even fp32 -> bf16
  unsigned u = __float_as_uint(f);
  unsigned r = (u + 0x7FFFu + ((u >> 16) & 1u)) >> 16;
  unsigned short h = (unsigned short)r;
  return __builtin_bit_cast(__bf16, h);
}

__device__ __forceinline__ v8f v8f_zero() {
  v8f z;
#pragma unroll
  for (int i = 0; i < 8; ++i) z[i] = 0.0f;
  return z;
}

// Low 32 bits of a generic pointer to LDS == wave-relative LDS byte offset
// (flat aperture).
__device__ __forceinline__ unsigned lds_off(const void* p) {
  return (unsigned)(size_t)p;
}

// gfx1250 async copy: 16B per lane, global -> LDS, tracked by ASYNCcnt.
__device__ __forceinline__ void async_copy_b128(unsigned ldsDst, const void* gsrc) {
  asm volatile("global_load_async_to_lds_b128 %0, %1, off"
               :: "v"(ldsDst), "v"((unsigned long long)(size_t)gsrc)
               : "memory");
}
__device__ __forceinline__ void wait_async() {
  asm volatile("s_wait_asynccnt 0x0" ::: "memory");
}

// ---------------------------------------------------------------------------
// Kernel 0: fp32 -> bf16 elementwise (n multiple of 1024).
// ---------------------------------------------------------------------------
__global__ __launch_bounds__(256) void cvt_bf16(const float* __restrict__ in,
                                                __bf16* __restrict__ out, int n) {
  const int i = (blockIdx.x * 256 + threadIdx.x) * 4;
  if (i < n) {
    float4 f = *(const float4*)(in + i);
    out[i + 0] = f2bf(f.x);
    out[i + 1] = f2bf(f.y);
    out[i + 2] = f2bf(f.z);
    out[i + 3] = f2bf(f.w);
  }
}

// ---------------------------------------------------------------------------
// Kernel 1: C = A[M,K](bf16) @ W[N,K](bf16)^T + bias, fp32 accumulate.
//   mode 0: fp32 C[M,N] (scaled)
//   mode 1: bf16 scatter into (B,H,S,HD)            (Q with scale, K)
//   mode 2: bf16 scatter into (B,H,HD,S) transposed (V)
// grid=(N/128, M/128), block=256 (8 waves, 2x4 wave grid, 64x32 per wave).
// Async double-buffered 128x32 LDS tiles.
// ---------------------------------------------------------------------------
__global__ __launch_bounds__(256) void gemm_bf16_wmma(
    const __bf16* __restrict__ A, const __bf16* __restrict__ W,
    const float* __restrict__ bias, void* __restrict__ Cout,
    int M, int N, int K, int mode, float outScale)
{
  __shared__ __align__(16) __bf16 As[2][128][40];  // 128x32 bf16, padded rows
  __shared__ __align__(16) __bf16 Ws[2][128][40];

  const int tid  = threadIdx.x;
  const int lane = tid & 31, wave = tid >> 5;
  const int l16  = lane & 15, hf = lane >> 4;
  const int m0 = blockIdx.y * 128;
  const int n0 = blockIdx.x * 128;
  const int mW = (wave & 1) * 64;
  const int nW = (wave >> 1) * 32;

  const int ldRow = tid >> 1;        // 0..127
  const int ldC0  = (tid & 1) * 2;   // 16B-chunk base within row (0 or 2)

  auto issue = [&](int kt, int buf) {
#pragma unroll
    for (int i = 0; i < 2; ++i) {
      const int c = ldC0 + i;        // chunk 0..3, 8 bf16 each
      async_copy_b128(lds_off(&As[buf][ldRow][c * 8]),
                      A + (size_t)(m0 + ldRow) * K + kt + c * 8);
      async_copy_b128(lds_off(&Ws[buf][ldRow][c * 8]),
                      W + (size_t)(n0 + ldRow) * K + kt + c * 8);
    }
  };

  v8f acc[4][2];
#pragma unroll
  for (int i = 0; i < 4; ++i)
#pragma unroll
    for (int j = 0; j < 2; ++j) acc[i][j] = v8f_zero();

  issue(0, 0);
  wait_async();
  __syncthreads();

  int buf = 0;
  for (int kt = 0; kt < K; kt += 32) {
    if (kt + 32 < K) issue(kt + 32, buf ^ 1);   // prefetch next tile

    // A fragments: contiguous K order, lane half selects K 0-15/16-31 (matches B).
    v16bf afr[4];
#pragma unroll
    for (int mf = 0; mf < 4; ++mf) {
      union { v16bf v; unsigned u[8]; } t;
      const int mr = mW + mf * 16 + l16;
#pragma unroll
      for (int vv = 0; vv < 8; ++vv) {
        const int k0 = (hf << 4) + (vv << 1);
        t.u[vv] = *(const unsigned*)&As[buf][mr][k0];
      }
      afr[mf] = t.v;
    }
    // B fragments: contiguous K order, same as A (lane half selects K 0-15/16-31).
    v16bf bfr[2];
#pragma unroll
    for (int nf = 0; nf < 2; ++nf) {
      union { v16bf v; float4 q[2]; } t;
      const __bf16* p = &Ws[buf][nW + nf * 16 + l16][hf << 4];
      t.q[0] = *(const float4*)p;
      t.q[1] = *(const float4*)(p + 8);
      bfr[nf] = t.v;
    }
#pragma unroll
    for (int mf = 0; mf < 4; ++mf)
#pragma unroll
      for (int nf = 0; nf < 2; ++nf)
        acc[mf][nf] = __builtin_amdgcn_wmma_f32_16x16x32_bf16(
            false, afr[mf], false, bfr[nf], (short)0, acc[mf][nf], false, false);

    wait_async();
    __syncthreads();
    buf ^= 1;
  }

  // Epilogue.
  float*  Cf = (float*)Cout;
  __bf16* Cb = (__bf16*)Cout;
#pragma unroll
  for (int mf = 0; mf < 4; ++mf) {
#pragma unroll
    for (int nf = 0; nf < 2; ++nf) {
      const int n = n0 + nW + nf * 16 + l16;
      const float bz = bias[n];
#pragma unroll
      for (int r = 0; r < 8; ++r) {
        const int m = m0 + mW + mf * 16 + r + hf * 8;
        const float val = (acc[mf][nf][r] + bz) * outScale;
        if (mode == 0) {
          Cf[(size_t)m * N + n] = val;
        } else {
          const int b = m >> 11, s = m & (S_LEN - 1);
          const int h = n >> 6,  hd = n & (HDIM - 1);
          if (mode == 1)
            Cb[(((size_t)b * NHEAD + h) * S_LEN + s) * HDIM + hd] = f2bf(val);
          else  // mode 2: V transposed (B,H,HD,S)
            Cb[(((size_t)b * NHEAD + h) * HDIM + hd) * S_LEN + s] = f2bf(val);
        }
      }
    }
  }
}

// ---------------------------------------------------------------------------
// Kernel 2: flash attention, causal + rel_bias, bf16 WMMA, async K/V
// double buffering. Q pre-scaled by 1/sqrt(HD); V pre-transposed (B,H,HD,S).
// grid=(S/128, B*H), block=256. Wave w owns q rows [qBase+16w, +16).
// Writes bf16 attended in (B,S,D) layout.
// ---------------------------------------------------------------------------
__global__ __launch_bounds__(256) void attn_flash_wmma(
    const __bf16* __restrict__ Q, const __bf16* __restrict__ Km,
    const __bf16* __restrict__ Vt_g, const float* __restrict__ relb,
    __bf16* __restrict__ att)
{
  __shared__ __align__(16) __bf16 Qs[128][72];     // 128x64 Q tile
  __shared__ __align__(16) __bf16 Ks[2][32][72];   // K blocks (double buffer)
  __shared__ __align__(16) __bf16 Vt[2][64][40];   // V^T blocks [hd][k]
  __shared__ __align__(16) __bf16 Ps[8][16][40];   // per-wave P scratch 16x32

  const int tid  = threadIdx.x;
  const int lane = tid & 31, wave = tid >> 5;
  const int l16  = lane & 15, hf = lane >> 4;
  const int qBase = blockIdx.x * 128;
  const int bh = blockIdx.y;
  const int b = bh >> 4, h = bh & 15;

  const __bf16* Qp = Q    + (size_t)bh * S_LEN * HDIM;
  const __bf16* Kp = Km   + (size_t)bh * S_LEN * HDIM;
  const __bf16* Vp = Vt_g + (size_t)bh * HDIM * S_LEN;
  const float*  Rp = relb + (size_t)h * S_LEN * S_LEN;

  // Q tile: 128 rows x 128B = 1024 chunks -> 4 async copies per thread.
  {
    const int row = tid >> 1;
    const int cb  = (tid & 1) * 4;
#pragma unroll
    for (int i = 0; i < 4; ++i)
      async_copy_b128(lds_off(&Qs[row][(cb + i) * 8]),
                      Qp + (size_t)(qBase + row) * HDIM + (cb + i) * 8);
  }

  auto issueKV = [&](int kb, int buf) {
    const int krow = tid >> 3, kc = tid & 7;   // 32 rows x 8 chunks
    async_copy_b128(lds_off(&Ks[buf][krow][kc * 8]),
                    Kp + (size_t)(kb + krow) * HDIM + kc * 8);
    const int vrow = tid >> 2, vc = tid & 3;   // 64 hd-rows x 4 chunks
    async_copy_b128(lds_off(&Vt[buf][vrow][vc * 8]),
                    Vp + (size_t)vrow * S_LEN + kb + vc * 8);
  };

  issueKV(0, 0);
  wait_async();
  __syncthreads();

  const int qw = qBase + wave * 16;

  // Per-wave Q A-fragments (hd chunks 0-31 / 32-63), constant over k loop.
  v16bf qa[2];
#pragma unroll
  for (int c = 0; c < 2; ++c) {
    union { v16bf v; unsigned u[8]; } t;
#pragma unroll
    for (int vv = 0; vv < 8; ++vv) {
      const int k0 = c * 32 + (hf << 4) + (vv << 1);
      t.u[vv] = *(const unsigned*)&Qs[wave * 16 + l16][k0];
    }
    qa[c] = t.v;
  }

  float mrow[8], lrow[8];
  v8f o[4];
#pragma unroll
  for (int r = 0; r < 8; ++r) { mrow[r] = -1e30f; lrow[r] = 0.0f; }
#pragma unroll
  for (int t4 = 0; t4 < 4; ++t4) o[t4] = v8f_zero();

  const int kEnd = qBase + 128;          // causal: blocks up to the diagonal
  int buf = 0;
  for (int kb = 0; kb < kEnd; kb += 32) {
    if (kb + 32 < kEnd) issueKV(kb + 32, buf ^ 1);   // prefetch next block

    if (kb <= qw + 15) {                 // wave has unmasked work in block
      // ---- scores: two 16-wide k sub-tiles ----
      v8f sc[2];
#pragma unroll
      for (int ks2 = 0; ks2 < 2; ++ks2) {
        union { v16bf v; float4 q[2]; } t0, t1;
        const __bf16* p0 = &Ks[buf][ks2 * 16 + l16][hf << 4];
        t0.q[0] = *(const float4*)p0;  t0.q[1] = *(const float4*)(p0 + 8);
        const __bf16* p1 = &Ks[buf][ks2 * 16 + l16][32 + (hf << 4)];
        t1.q[0] = *(const float4*)p1;  t1.q[1] = *(const float4*)(p1 + 8);
        v8f z = v8f_zero();
        z = __builtin_amdgcn_wmma_f32_16x16x32_bf16(false, qa[0], false, t0.v,
                                                    (short)0, z, false, false);
        z = __builtin_amdgcn_wmma_f32_16x16x32_bf16(false, qa[1], false, t1.v,
                                                    (short)0, z, false, false);
        sc[ks2] = z;
      }

      // ---- rel_bias + causal mask + online softmax ----
#pragma unroll
      for (int r = 0; r < 8; ++r) {
        const int q   = qw + r + hf * 8;
        const int k0g = kb + l16;
        float s0 = sc[0][r], s1 = sc[1][r];
        s0 = (k0g      <= q) ? s0 + Rp[(size_t)q * S_LEN + k0g]      : -1e30f;
        s1 = (k0g + 16 <= q) ? s1 + Rp[(size_t)q * S_LEN + k0g + 16] : -1e30f;
        float lm = fmaxf(s0, s1);
#pragma unroll
        for (int off = 8; off >= 1; off >>= 1)
          lm = fmaxf(lm, __shfl_xor(lm, off, 32));
        const float mn  = fmaxf(mrow[r], lm);
        const float scl = __expf(mrow[r] - mn);
        mrow[r] = mn;
        const float e0 = __expf(s0 - mn), e1 = __expf(s1 - mn);
        float rs = e0 + e1;
#pragma unroll
        for (int off = 8; off >= 1; off >>= 1)
          rs += __shfl_xor(rs, off, 32);
        lrow[r] = lrow[r] * scl + rs;
#pragma unroll
        for (int t4 = 0; t4 < 4; ++t4) o[t4][r] *= scl;
        Ps[wave][r + hf * 8][l16]      = f2bf(e0);
        Ps[wave][r + hf * 8][l16 + 16] = f2bf(e1);
      }

      // ---- O += P @ V ----
      v16bf pa;
      {
        union { v16bf v; unsigned u[8]; } t;
#pragma unroll
        for (int vv = 0; vv < 8; ++vv) {
          const int k0 = (hf << 4) + (vv << 1);
          t.u[vv] = *(const unsigned*)&Ps[wave][l16][k0];
        }
        pa = t.v;
      }
#pragma unroll
      for (int t4 = 0; t4 < 4; ++t4) {
        union { v16bf v; float4 q[2]; } tb;
        const __bf16* p = &Vt[buf][t4 * 16 + l16][hf << 4];
        tb.q[0] = *(const float4*)p;  tb.q[1] = *(const float4*)(p + 8);
        o[t4] = __builtin_amdgcn_wmma_f32_16x16x32_bf16(false, pa, false, tb.v,
                                                        (short)0, o[t4], false, false);
      }
    }

    wait_async();
    __syncthreads();
    buf ^= 1;
  }

  // normalize, write bf16 attended in (B,S,D)
#pragma unroll
  for (int t4 = 0; t4 < 4; ++t4) {
#pragma unroll
    for (int r = 0; r < 8; ++r) {
      const int q = qw + r + hf * 8;
      const float inv = 1.0f / lrow[r];
      att[((size_t)b * S_LEN + q) * DMODEL + h * HDIM + t4 * 16 + l16] =
          f2bf(o[t4][r] * inv);
    }
  }
}

// ---------------------------------------------------------------------------
// Kernel 3: y = proj + x ; LayerNorm(eps=1e-5) * gamma + beta. One WG per row.
// ---------------------------------------------------------------------------
__global__ __launch_bounds__(256) void resid_layernorm(
    const float* __restrict__ proj, const float* __restrict__ x,
    const float* __restrict__ gamma, const float* __restrict__ beta,
    float* __restrict__ out)
{
  const int row  = blockIdx.x;
  const int tid  = threadIdx.x;
  const int lane = tid & 31, wave = tid >> 5;
  const size_t base = (size_t)row * DMODEL;

  float v[4]; float s = 0.f, ss = 0.f;
#pragma unroll
  for (int i = 0; i < 4; ++i) {
    const int idx = tid + i * 256;
    v[i] = proj[base + idx] + x[base + idx];
    s += v[i]; ss += v[i] * v[i];
  }
#pragma unroll
  for (int off = 16; off >= 1; off >>= 1) {
    s  += __shfl_xor(s,  off, 32);
    ss += __shfl_xor(ss, off, 32);
  }
  __shared__ float rs[8], rss[8], stats[2];
  if (lane == 0) { rs[wave] = s; rss[wave] = ss; }
  __syncthreads();
  if (tid == 0) {
    float S = 0.f, SS = 0.f;
#pragma unroll
    for (int i = 0; i < 8; ++i) { S += rs[i]; SS += rss[i]; }
    const float mu  = S / (float)DMODEL;
    const float var = SS / (float)DMODEL - mu * mu;
    stats[0] = mu;
    stats[1] = rsqrtf(var + 1e-5f);
  }
  __syncthreads();
  const float mu = stats[0], rstd = stats[1];
#pragma unroll
  for (int i = 0; i < 4; ++i) {
    const int idx = tid + i * 256;
    out[base + idx] = (v[i] - mu) * rstd * gamma[idx] + beta[idx];
  }
}

// ---------------------------------------------------------------------------
extern "C" void kernel_launch(void* const* d_in, const int* in_sizes, int n_in,
                              void* d_out, int out_size, void* d_ws, size_t ws_size,
                              hipStream_t stream) {
  const float* x     = (const float*)d_in[0];
  const float* Wq    = (const float*)d_in[1];
  const float* bq    = (const float*)d_in[2];
  const float* Wk    = (const float*)d_in[3];
  const float* bk    = (const float*)d_in[4];
  const float* Wv    = (const float*)d_in[5];
  const float* bv    = (const float*)d_in[6];
  const float* Wo    = (const float*)d_in[7];
  const float* bo    = (const float*)d_in[8];
  const float* relb  = (const float*)d_in[9];
  const float* gamma = (const float*)d_in[10];
  const float* beta  = (const float*)d_in[11];

  const int M = BATCH * S_LEN;                 // 4096
  const int N = DMODEL, K = DMODEL;            // 1024
  const size_t mdElems = (size_t)M * DMODEL;   // 4M
  const size_t wElems  = (size_t)DMODEL * DMODEL;

  char* p = (char*)d_ws;
  __bf16* xb   = (__bf16*)p; p += mdElems * 2;
  __bf16* Wqb  = (__bf16*)p; p += wElems * 2;
  __bf16* Wkb  = (__bf16*)p; p += wElems * 2;
  __bf16* Wvb  = (__bf16*)p; p += wElems * 2;
  __bf16* Wob  = (__bf16*)p; p += wElems * 2;
  __bf16* Qb   = (__bf16*)p; p += mdElems * 2;
  __bf16* Kb   = (__bf16*)p; p += mdElems * 2;
  __bf16* Vb   = (__bf16*)p; p += mdElems * 2;
  __bf16* attb = (__bf16*)p; p += mdElems * 2;
  float*  prj  = (float*)p;

  dim3 blk(256);

  // fp32 -> bf16 pre-pass (x + 4 weights)
  cvt_bf16<<<(int)(mdElems / 1024), blk, 0, stream>>>(x,  xb,  (int)mdElems);
  cvt_bf16<<<(int)(wElems  / 1024), blk, 0, stream>>>(Wq, Wqb, (int)wElems);
  cvt_bf16<<<(int)(wElems  / 1024), blk, 0, stream>>>(Wk, Wkb, (int)wElems);
  cvt_bf16<<<(int)(wElems  / 1024), blk, 0, stream>>>(Wv, Wvb, (int)wElems);
  cvt_bf16<<<(int)(wElems  / 1024), blk, 0, stream>>>(Wo, Wob, (int)wElems);

  dim3 gG(N / 128, M / 128);
  // Q (pre-scaled by 1/sqrt(HD)), K in (B,H,S,HD); V transposed (B,H,HD,S)
  gemm_bf16_wmma<<<gG, blk, 0, stream>>>(xb, Wqb, bq, Qb, M, N, K, 1, 0.125f);
  gemm_bf16_wmma<<<gG, blk, 0, stream>>>(xb, Wkb, bk, Kb, M, N, K, 1, 1.0f);
  gemm_bf16_wmma<<<gG, blk, 0, stream>>>(xb, Wvb, bv, Vb, M, N, K, 2, 1.0f);

  // flash attention with causal mask + relative position bias
  dim3 gA(S_LEN / 128, BATCH * NHEAD);
  attn_flash_wmma<<<gA, blk, 0, stream>>>(Qb, Kb, Vb, relb, attb);

  // output projection (fp32 out)
  gemm_bf16_wmma<<<gG, blk, 0, stream>>>(attb, Wob, bo, prj, M, N, K, 0, 1.0f);

  // residual + LayerNorm
  resid_layernorm<<<M, blk, 0, stream>>>(prj, x, gamma, beta, (float*)d_out);
}
